// CAFM_19954418057896
// MI455X (gfx1250) — matrix operand order repaired and checked by reading, not compiled, not run on hardware
//
#include <hip/hip_runtime.h>

typedef __attribute__((ext_vector_type(16))) _Float16 v16h;
typedef __attribute__((ext_vector_type(8)))  _Float16 v8h;
typedef __attribute__((ext_vector_type(8)))  float    v8f;
typedef int v4i __attribute__((vector_size(16)));

#define CDIM 256
#define NSEQ 2048
#define BATCH 16

#if defined(__gfx1250__) && __has_builtin(__builtin_amdgcn_global_load_async_to_lds_b128)
#define HAVE_ASYNC_LDS 1
#else
#define HAVE_ASYNC_LDS 0
#endif

__device__ __forceinline__ void async_wait0() {
#if HAVE_ASYNC_LDS
#if __has_builtin(__builtin_amdgcn_s_wait_asynccnt)
  __builtin_amdgcn_s_wait_asynccnt(0);
#else
  asm volatile("s_wait_asynccnt 0" ::: "memory");
#endif
#endif
}

__device__ __forceinline__ v16h pack16(v8h lo, v8h hi) {
  v16h r;
#pragma unroll
  for (int i = 0; i < 8; ++i) { r[i] = lo[i]; r[8 + i] = hi[i]; }
  return r;
}

__device__ __forceinline__ v8f wmma32(v16h a, v16h b, v8f c) {
  return __builtin_amdgcn_wmma_f32_16x16x32_f16(false, a, false, b,
                                                (short)0, c, false, false);
}

// A fragment (16x32 f16): lane = row (lane&15); K halves per ISA layout:
// lanes 0-15 hold K {0..7, 16..23}, lanes 16-31 hold K {8..15, 24..31}.
__device__ __forceinline__ v16h ldA16x32(const _Float16* base, int ld, int lane) {
  int m  = lane & 15;
  int kh = (lane >> 4) << 3;                    // 0 or 8
  const _Float16* p = base + m * ld + kh;
  v8h lo = *(const v8h*)p;                      // K = kh .. kh+7
  v8h hi = *(const v8h*)(p + 16);               // K = kh+16 .. kh+23
  return pack16(lo, hi);
}

// B fragment (32x16 f16): lane = column n (lane&15); K contiguous within a row
// of `base`; lanes 0-15: K {0..7, 16..23}, lanes 16-31: K {8..15, 24..31}.
__device__ __forceinline__ v16h ldB32x16(const _Float16* base, int ld, int lane) {
  int n  = lane & 15;
  int ko = (lane >> 4) << 3;                    // 0 or 8
  const _Float16* p = base + n * ld + ko;
  v8h lo = *(const v8h*)p;
  v8h hi = *(const v8h*)(p + 16);
  return pack16(lo, hi);
}

// ---------------------------------------------------------------------------
// Projection GEMM: O[m,n] = X[m,:] @ W[:,n] + bias, f32 in -> f16 out.
// X logical [32768, 256] but stored as feat[b, c, n] (seq transposed).
// grid = (M/64, N/64, 3), block = 128 (4 waves). z selects Q/K/V projection.
// ---------------------------------------------------------------------------
#define PLD 40   // padded LDS leading dim (multiple of 8)

__global__ __launch_bounds__(128)
void cafm_proj_kernel(const float* __restrict__ xq, const float* __restrict__ xkv,
                      const float* __restrict__ w0, const float* __restrict__ b0,
                      const float* __restrict__ w1, const float* __restrict__ b1,
                      const float* __restrict__ w2, const float* __restrict__ b2,
                      _Float16* __restrict__ oq, _Float16* __restrict__ ok,
                      _Float16* __restrict__ ov)
{
  __shared__ _Float16 Xs[64 * PLD];   // [row m][k]        (f16)
  __shared__ _Float16 Wt[64 * PLD];   // [col n][k] transposed (f16)

  const int z = blockIdx.z;
  const float* X  = (z == 0) ? xq : xkv;
  const float* W  = (z == 0) ? w0 : (z == 1 ? w1 : w2);
  const float* Bi = (z == 0) ? b0 : (z == 1 ? b1 : b2);
  _Float16*    O  = (z == 0) ? oq : (z == 1 ? ok : ov);

  const int tid  = threadIdx.x;
  const int lane = tid & 31;
  const int wv   = tid >> 5;

  const int mBase = blockIdx.x * 64;
  const int nBase = blockIdx.y * 64;
  const int batch = mBase / NSEQ;     // 64 | 2048, tiles never straddle batches
  const int n0    = mBase % NSEQ;

  v8f acc[4] = {};

  for (int kb = 0; kb < CDIM; kb += 32) {
    // X tile: gather feat[b, kb+cc, n0+m] -> Xs[m][cc]; coalesced along m.
#pragma unroll
    for (int i = 0; i < 16; ++i) {
      int idx = tid + i * 128;                  // 0..2047
      int m   = idx & 63;
      int cc  = idx >> 6;
      float v = X[(size_t)batch * CDIM * NSEQ + (size_t)(kb + cc) * NSEQ + n0 + m];
      Xs[m * PLD + cc] = (_Float16)v;
    }
    // W tile: W[kb+r][nBase+c] -> Wt[c][r]; coalesced along c.
#pragma unroll
    for (int i = 0; i < 16; ++i) {
      int idx = tid + i * 128;
      int c   = idx & 63;
      int r   = idx >> 6;
      float v = W[(size_t)(kb + r) * CDIM + nBase + c];
      Wt[c * PLD + r] = (_Float16)v;
    }
    __syncthreads();

    v16h a = ldA16x32(Xs + (wv * 16) * PLD, PLD, lane);
#pragma unroll
    for (int t = 0; t < 4; ++t) {
      v16h b = ldB32x16(Wt + (t * 16) * PLD, PLD, lane);
      acc[t] = wmma32(a, b, acc[t]);
    }
    __syncthreads();
  }

  const int nn   = lane & 15;
  const int mOff = (lane >> 4) << 3;
#pragma unroll
  for (int t = 0; t < 4; ++t) {
    float bb = Bi[nBase + t * 16 + nn];
#pragma unroll
    for (int r = 0; r < 8; ++r) {
      int row = mBase + wv * 16 + mOff + r;
      O[(size_t)row * CDIM + nBase + t * 16 + nn] = (_Float16)(acc[t][r] + bb);
    }
  }
}

// ---------------------------------------------------------------------------
// Flash attention + mean-over-queries.  grid = 256 (16 q-blocks x 16 batches),
// block = 256 (8 waves, 16 query rows each). Double-buffered K/V tiles in LDS;
// K tiles copied with GLOBAL_LOAD_ASYNC_TO_LDS (ASYNCcnt) when available so the
// next tile streams in while WMMAs run on the current one. Accumulates
// mean_n(O[b,n,c]) into fused[b][outOff + c] with float atomics.
// ---------------------------------------------------------------------------
#define LDK 264   // K tile leading dim (256 + 8 pad)
#define LDV 40    // V^T tile leading dim (32 + 8 pad)

__global__ __launch_bounds__(256)
void cafm_flash_kernel(const _Float16* __restrict__ Q, const _Float16* __restrict__ K,
                       const _Float16* __restrict__ V, float* __restrict__ fused,
                       int outOff)
{
  __shared__ _Float16 Ks[2][32 * LDK];     // [buf][key][chan]
  __shared__ _Float16 Vt[2][CDIM * LDV];   // [buf][chan][key]  (transposed)
  __shared__ _Float16 Ps[8 * 16 * 32];     // per-wave P scratch [16 rows][32 keys]

  const int tid  = threadIdx.x;
  const int lane = tid & 31;
  const int wv   = tid >> 5;

  const int batch = blockIdx.x >> 4;
  const int qBlk  = blockIdx.x & 15;
  const int qRow0 = qBlk * 128 + wv * 16;  // within batch

  const _Float16* Qb = Q + (size_t)batch * NSEQ * CDIM;
  const _Float16* Kb = K + (size_t)batch * NSEQ * CDIM;
  const _Float16* Vb = V + (size_t)batch * NSEQ * CDIM;

  // Issue one 32-key x 256-chan tile copy into LDS buffer `buf`.
  // K tile: row-major, 16B-granular async DMA to LDS when available.
  // V tile: register round-trip with element transpose into Vt[buf].
  auto loadTile = [&](int buf, int kb) {
#pragma unroll
    for (int i = 0; i < 4; ++i) {
      int idx = tid + i * 256;             // 0..1023 (v8h chunks)
      int row = idx >> 5;
      int c8  = (idx & 31) << 3;
      const _Float16* gk = Kb + (size_t)(kb + row) * CDIM + c8;
#if HAVE_ASYNC_LDS
      __builtin_amdgcn_global_load_async_to_lds_b128(
          (v4i __attribute__((address_space(1)))*)gk,
          (v4i __attribute__((address_space(3)))*)&Ks[buf][row * LDK + c8],
          0, 0);
#else
      *(v8h*)&Ks[buf][row * LDK + c8] = *(const v8h*)gk;
#endif
      v8h vvec = *(const v8h*)(Vb + (size_t)(kb + row) * CDIM + c8);
#pragma unroll
      for (int j = 0; j < 8; ++j) Vt[buf][(c8 + j) * LDV + row] = vvec[j];
    }
  };

  // Q fragments held in registers (16 rows x 256 channels = 8 K-chunks).
  v16h qf[8];
  {
    int m  = lane & 15;
    int kh = (lane >> 4) << 3;
    const _Float16* qrow = Qb + (size_t)(qRow0 + m) * CDIM;
#pragma unroll
    for (int cb = 0; cb < 8; ++cb) {
      v8h lo = *(const v8h*)(qrow + cb * 32 + kh);
      v8h hi = *(const v8h*)(qrow + cb * 32 + kh + 16);
      qf[cb] = pack16(lo, hi);
    }
  }

  v8f acc[16] = {};                        // O accumulators: 16 col tiles
  float mrow[8], lrow[8];
#pragma unroll
  for (int r = 0; r < 8; ++r) { mrow[r] = -1e30f; lrow[r] = 0.0f; }

  _Float16* Pw = Ps + wv * (16 * 32);
  const float scale = 0.0625f;             // 1/sqrt(256)
  const int nn = lane & 15;

  // Prologue: stage tile 0 into buffer 0.
  loadTile(0, 0);
  async_wait0();
  __syncthreads();

  for (int it = 0; it < NSEQ / 32; ++it) {
    const int cur = it & 1;
    const int kb  = it * 32;

    // Prefetch next tile into the other buffer (no barrier needed: nobody
    // reads it this iteration; the end-of-iteration barrier publishes it).
    if (it + 1 < NSEQ / 32) loadTile(cur ^ 1, kb + 32);

    // Scores: two 16x16 tiles (key halves), 8 K-chunks each.
    v8f s0 = {}, s1 = {};
#pragma unroll
    for (int cb = 0; cb < 8; ++cb) {
      v16h b0 = ldB32x16(&Ks[cur][cb * 32], LDK, lane);            // keys 0..15
      v16h b1 = ldB32x16(&Ks[cur][16 * LDK + cb * 32], LDK, lane); // keys 16..31
      s0 = wmma32(qf[cb], b0, s0);
      s1 = wmma32(qf[cb], b1, s1);
    }

    // Online softmax (rows live per VGPR slice; reduce across 16-lane halves).
#pragma unroll
    for (int r = 0; r < 8; ++r) {
      float a0 = s0[r] * scale;
      float a1 = s1[r] * scale;
      float tmax = fmaxf(a0, a1);
#pragma unroll
      for (int off = 1; off < 16; off <<= 1)
        tmax = fmaxf(tmax, __shfl_xor(tmax, off, 32));
      float mn   = fmaxf(mrow[r], tmax);
      float corr = __expf(mrow[r] - mn);
      mrow[r] = mn;
      float p0 = __expf(a0 - mn);
      float p1 = __expf(a1 - mn);
      float rs = p0 + p1;
#pragma unroll
      for (int off = 1; off < 16; off <<= 1)
        rs += __shfl_xor(rs, off, 32);
      lrow[r] = lrow[r] * corr + rs;
#pragma unroll
      for (int t = 0; t < 16; ++t) acc[t][r] *= corr;
      int row = ((lane >> 4) << 3) + r;
      Pw[row * 32 + nn]      = (_Float16)p0;   // keys 0..15
      Pw[row * 32 + 16 + nn] = (_Float16)p1;   // keys 16..31
    }

    // P @ V: A frag from wave-private LDS (DS ops are in-order per wave),
    // B frags from the transposed V tile.
    v16h ap = ldA16x32(Pw, 32, lane);
#pragma unroll
    for (int t = 0; t < 16; ++t) {
      v16h bv = ldB32x16(&Vt[cur][(t * 16) * LDV], LDV, lane);
      acc[t] = wmma32(ap, bv, acc[t]);
    }

    // Drain this wave's async prefetch, then publish buffers to the WG.
    async_wait0();
    __syncthreads();
  }

  // Normalize rows, reduce over the 16 query rows, accumulate column means.
  float invl[8];
#pragma unroll
  for (int r = 0; r < 8; ++r) invl[r] = 1.0f / lrow[r];
  const float wgt = 1.0f / (float)NSEQ;
#pragma unroll
  for (int t = 0; t < 16; ++t) {
    float v = 0.0f;
#pragma unroll
    for (int r = 0; r < 8; ++r) v += acc[t][r] * invl[r];
    v += __shfl_xor(v, 16, 32);              // combine rows 0-7 and 8-15 halves
    if (lane < 16)
      atomicAdd(&fused[batch * 512 + outOff + t * 16 + nn], v * wgt);
  }
}

// ---------------------------------------------------------------------------
// Head: out = relu(fused @ W1 + b1) @ W2 + b2.  grid = 16, block = 256.
// ---------------------------------------------------------------------------
__global__ __launch_bounds__(256)
void cafm_head_kernel(const float* __restrict__ fused,
                      const float* __restrict__ w1, const float* __restrict__ b1,
                      const float* __restrict__ w2, const float* __restrict__ b2,
                      float* __restrict__ out)
{
  __shared__ float hid[256];
  const int b = blockIdx.x;
  const int c = threadIdx.x;
  float a = b1[c];
  for (int k = 0; k < 512; ++k) a += fused[b * 512 + k] * w1[k * 256 + c];
  hid[c] = fmaxf(a, 0.0f);
  __syncthreads();
  float o = b2[c];
  for (int k = 0; k < 256; ++k) o += hid[k] * w2[k * 256 + c];
  out[b * 256 + c] = o;
}

__global__ void cafm_zero_kernel(float* p, int n) {
  int i = blockIdx.x * 256 + threadIdx.x;
  if (i < n) p[i] = 0.0f;
}

// ---------------------------------------------------------------------------
extern "C" void kernel_launch(void* const* d_in, const int* in_sizes, int n_in,
                              void* d_out, int out_size, void* d_ws, size_t ws_size,
                              hipStream_t stream)
{
  (void)in_sizes; (void)n_in; (void)out_size; (void)ws_size;
  const float* stft = (const float*)d_in[0];
  const float* cqt  = (const float*)d_in[1];
  const float* w1w = (const float*)d_in[2];  const float* w1b = (const float*)d_in[3];
  const float* w2w = (const float*)d_in[4];  const float* w2b = (const float*)d_in[5];
  const float* w3w = (const float*)d_in[6];  const float* w3b = (const float*)d_in[7];
  const float* w4w = (const float*)d_in[8];  const float* w4b = (const float*)d_in[9];
  const float* w5w = (const float*)d_in[10]; const float* w5b = (const float*)d_in[11];
  const float* w6w = (const float*)d_in[12]; const float* w6b = (const float*)d_in[13];
  const float* o1w = (const float*)d_in[14]; const float* o1b = (const float*)d_in[15];
  const float* o2w = (const float*)d_in[16]; const float* o2b = (const float*)d_in[17];

  char* ws = (char*)d_ws;
  float* fused = (float*)ws;                               // [16][512] f32
  size_t off = ((size_t)BATCH * 512 * sizeof(float) + 255) & ~(size_t)255;
  const size_t projBytes = (size_t)BATCH * NSEQ * CDIM * sizeof(_Float16);
  _Float16* Qb = (_Float16*)(ws + off); off += projBytes;
  _Float16* Kb = (_Float16*)(ws + off); off += projBytes;
  _Float16* Vb = (_Float16*)(ws + off);

  cafm_zero_kernel<<<32, 256, 0, stream>>>(fused, BATCH * 512);

  dim3 pg((BATCH * NSEQ) / 64, CDIM / 64, 3), pb(128);
  // Attention 1: Q from stft, K/V from cqt.
  cafm_proj_kernel<<<pg, pb, 0, stream>>>(stft, cqt, w1w, w1b, w2w, w2b, w3w, w3b,
                                          Qb, Kb, Vb);
  cafm_flash_kernel<<<BATCH * (NSEQ / 128), 256, 0, stream>>>(Qb, Kb, Vb, fused, 0);
  // Attention 2: Q from cqt, K/V from stft.
  cafm_proj_kernel<<<pg, pb, 0, stream>>>(cqt, stft, w4w, w4b, w5w, w5b, w6w, w6b,
                                          Qb, Kb, Vb);
  cafm_flash_kernel<<<BATCH * (NSEQ / 128), 256, 0, stream>>>(Qb, Kb, Vb, fused, 256);

  cafm_head_kernel<<<BATCH, 256, 0, stream>>>(fused, o1w, o1b, o2w, o2b, (float*)d_out);
}
